// EnhancedSignalFeatureExtractor_16045997817873
// MI455X (gfx1250) — matrix operand order repaired and checked by reading, not compiled, not run on hardware
//
#include <hip/hip_runtime.h>

typedef __attribute__((ext_vector_type(16))) __bf16 v16bf;
typedef __attribute__((ext_vector_type(8)))  __bf16 v8bf;
typedef __attribute__((ext_vector_type(8)))  float  v8f;
typedef __attribute__((ext_vector_type(4)))  unsigned int v4u;
typedef __attribute__((ext_vector_type(8)))  int v8i;
typedef __attribute__((ext_vector_type(4)))  int v4i;

static __device__ __forceinline__ unsigned short f32_to_bf16(float f) {
  unsigned int u = __float_as_uint(f);
  if ((u & 0x7fffffffu) > 0x7f800000u) return (unsigned short)((u >> 16) | 0x40); // quiet NaN
  u += 0x7fffu + ((u >> 16) & 1u);   // round-to-nearest-even
  return (unsigned short)(u >> 16);
}

// ---------------------------------------------------------------------------
// conv0 (K=7, pad=3, cin=6) + BN + ReLU + maxpool2 ; output NLC [32][2048][64]
// ---------------------------------------------------------------------------
__global__ void __launch_bounds__(256)
conv0_bn_relu_pool(const float* __restrict__ x, const float* __restrict__ w,
                   const float* __restrict__ b, const float* __restrict__ g,
                   const float* __restrict__ be, const float* __restrict__ mu,
                   const float* __restrict__ var, float* __restrict__ y, long total) {
  long t = (long)blockIdx.x * blockDim.x + threadIdx.x;
  if (t >= total) return;
  int c  = (int)(t % 64);
  long nl = t / 64;
  int lp = (int)(nl % 2048);
  int n  = (int)(nl / 2048);
  float scale = g[c] * rsqrtf(var[c] + 1e-5f);
  float shift = (b[c] - mu[c]) * scale + be[c];
  float m = -1e30f;
#pragma unroll
  for (int s = 0; s < 2; ++s) {
    int l0 = 2 * lp + s;
    float acc = 0.f;
    for (int i = 0; i < 6; ++i) {
      const float* xr = x + ((size_t)n * 6 + i) * 4096;
      const float* wr = w + ((size_t)c * 6 + i) * 7;
#pragma unroll
      for (int k = 0; k < 7; ++k) {
        int ll = l0 + k - 3;
        if (ll >= 0 && ll < 4096) acc += wr[k] * xr[ll];
      }
    }
    m = fmaxf(m, fmaxf(acc * scale + shift, 0.f));
  }
  y[((size_t)n * 2048 + lp) * 64 + c] = m;
}

// ---------------------------------------------------------------------------
// im2col (K=3, pad=1) NLC f32 -> [n][l][ci*3+k] bf16
// ---------------------------------------------------------------------------
__global__ void __launch_bounds__(256)
im2col3(const float* __restrict__ X, unsigned short* __restrict__ Xc,
        int L, int C, long total) {
  long t = (long)blockIdx.x * blockDim.x + threadIdx.x;
  if (t >= total) return;
  int ci = (int)(t % C);
  long nl = t / C;
  int l = (int)(nl % L);
  int n = (int)(nl / L);
  unsigned short* dst = Xc + ((size_t)n * L + l) * (3 * (size_t)C) + (size_t)ci * 3;
#pragma unroll
  for (int k = 0; k < 3; ++k) {
    int ll = l + k - 1;
    float v = (ll >= 0 && ll < L) ? X[((size_t)n * L + ll) * C + ci] : 0.f;
    dst[k] = f32_to_bf16(v);
  }
}

// ---------------------------------------------------------------------------
// Pack weights [M][Cin][kw] -> bf16 [M][Cin*3]; kw==1 lands in k=1 slot
// ---------------------------------------------------------------------------
__global__ void __launch_bounds__(256)
pack_weights(const float* __restrict__ W, unsigned short* __restrict__ Wp,
             int M, int Cin, int kw, long total) {
  long t = (long)blockIdx.x * blockDim.x + threadIdx.x;
  if (t >= total) return;
  int k  = (int)(t % 3);
  int ci = (int)((t / 3) % Cin);
  int m  = (int)(t / (3L * Cin));
  float v;
  if (kw == 3) v = W[((size_t)m * Cin + ci) * 3 + k];
  else         v = (k == 1) ? W[(size_t)m * Cin + ci] : 0.f;
  Wp[(size_t)m * (3 * (size_t)Cin) + (size_t)ci * 3 + k] = f32_to_bf16(v);
}

// Fold conv bias + (optional) BN into per-channel alpha/beta
__global__ void make_affine(const float* __restrict__ b, const float* __restrict__ g,
                            const float* __restrict__ be, const float* __restrict__ mu,
                            const float* __restrict__ var, float* __restrict__ alpha,
                            float* __restrict__ beta, int M, int has_bn) {
  int t = blockIdx.x * blockDim.x + threadIdx.x;
  if (t >= M) return;
  if (has_bn) {
    float s = g[t] * rsqrtf(var[t] + 1e-5f);
    alpha[t] = s;
    beta[t]  = (b[t] - mu[t]) * s + be[t];
  } else { alpha[t] = 1.f; beta[t] = b[t]; }
}

// ---------------------------------------------------------------------------
// TDM-staged WMMA GEMM (M multiple of 128):
// Workgroup = 8 waves sharing one 32-wide L panel; wave w owns M rows
// [w*16*MT, (w+1)*16*MT). The B panel is DMA'd to LDS in 96-element K chunks
// by TENSOR_LOAD_TO_LDS (double-buffered, wave 0 issues, s_wait_tensorcnt),
// with D# LDS padding of 4 dwords per 16 dwords so each 32-elem WMMA K-step
// sub-block has 80B stride (16B aligned, bank-friendly).
// ---------------------------------------------------------------------------
template <int MT>
__global__ void __launch_bounds__(256)
gemm_conv_wmma_tdm(const __bf16* __restrict__ Xc, const __bf16* __restrict__ Wp,
                   const float* __restrict__ alpha, const float* __restrict__ beta,
                   float* __restrict__ Y, int L, int M, int K, int relu) {
  // 96 sub-blocks of (32 data + 8 pad) bf16 per buffer
  __shared__ __align__(16) unsigned short panel[2][96 * 40];
  const int tid  = threadIdx.x;
  const int wv   = tid >> 5;
  const int lane = tid & 31;
  const int half = lane >> 4;
  const int lm   = lane & 15;
  const int lpb  = L >> 5;                      // 32-wide l panels per batch
  const int n    = blockIdx.x / lpb;
  const int lblk = (blockIdx.x - n * lpb) << 5;

  // A fragment: lane element j -> K = (j&8)*2 + half*8 + (j&7)
  const __bf16* wrow[MT];
#pragma unroll
  for (int mi = 0; mi < MT; ++mi)
    wrow[mi] = Wp + (size_t)((wv * MT + mi) * 16 + lm) * K + (half << 3);

  v8f zero = {};
  v8f acc[MT][2];
#pragma unroll
  for (int mi = 0; mi < MT; ++mi)
#pragma unroll
    for (int li = 0; li < 2; ++li) acc[mi][li] = zero;

  const unsigned ldsbase =
      (unsigned)(unsigned long long)(size_t)(void*)&panel[0][0];
  const unsigned long long gbase =
      (unsigned long long)(size_t)(const void*)(Xc + ((size_t)n * L + lblk) * K);
  const int nchunks = K / 96;

  auto issue = [&](int c) {
    unsigned long long ga = gbase + (unsigned long long)c * 192u; // 96 elems * 2B
    unsigned lds = ldsbase + (unsigned)(c & 1) * (96u * 40u * 2u);
    v4u g0 = { 1u,                                  // count=1
               lds,                                 // lds_addr
               (unsigned)ga,                        // global_addr[31:0]
               (unsigned)((ga >> 32) & 0x1ffffffu) | (2u << 30) }; // [56:32] | type=2
    v8i g1 = { (int)((1u << 16) | (1u << 20) | (3u << 22) | (3u << 25)),
               // data_size=2B, pad_enable, pad_interval=16dw, pad_amount=4dw
               (int)(96u << 16),                    // tensor_dim0 = 96
               (int)(32u << 16),                    // tensor_dim1 = 32
               (int)(96u << 16),                    // tile_dim0 = 96
               32,                                  // tile_dim1 = 32
               K,                                   // tensor_dim0_stride
               0, 0 };
    v4i gz = { 0, 0, 0, 0 };
    v8i gz8 = { 0, 0, 0, 0, 0, 0, 0, 0 };
    __builtin_amdgcn_tensor_load_to_lds(g0, g1, gz, gz, gz8, 0);
  };

  if (tid < 32) { issue(0); __builtin_amdgcn_s_wait_tensorcnt(0); }
  __syncthreads();

  for (int c = 0; c < nchunks; ++c) {
    if (tid < 32 && c + 1 < nchunks) issue(c + 1);
    const unsigned short* buf = panel[c & 1];
#pragma unroll
    for (int ks = 0; ks < 3; ++ks) {
      v16bf b[2];
#pragma unroll
      for (int li = 0; li < 2; ++li) {
        int row = li * 16 + lm;
        const __bf16* bp = (const __bf16*)(buf + (row * 3 + ks) * 40 + (half << 3));
        v8bf b0 = *(const v8bf*)bp;
        v8bf b1 = *(const v8bf*)(bp + 16);
#pragma unroll
        for (int j = 0; j < 8; ++j) { b[li][j] = b0[j]; b[li][j + 8] = b1[j]; }
      }
      const int kb = c * 96 + ks * 32;
#pragma unroll
      for (int mi = 0; mi < MT; ++mi) {
        v8bf a0 = *(const v8bf*)(wrow[mi] + kb);
        v8bf a1 = *(const v8bf*)(wrow[mi] + kb + 16);
        v16bf a;
#pragma unroll
        for (int j = 0; j < 8; ++j) { a[j] = a0[j]; a[j + 8] = a1[j]; }
#pragma unroll
        for (int li = 0; li < 2; ++li)
          acc[mi][li] = __builtin_amdgcn_wmma_f32_16x16x32_bf16(
              false, a, false, b[li], (short)0, acc[mi][li], false, false);
      }
    }
    if (tid < 32 && c + 1 < nchunks) __builtin_amdgcn_s_wait_tensorcnt(0);
    __syncthreads();
  }

#pragma unroll
  for (int mi = 0; mi < MT; ++mi) {
    const int cobase = (wv * MT + mi) * 16 + (half << 3);
#pragma unroll
    for (int li = 0; li < 2; ++li) {
      const int l = lblk + li * 16 + lm;
      float* yp = Y + ((size_t)n * L + l) * M + cobase;
#pragma unroll
      for (int rr = 0; rr < 8; ++rr) {
        float v = acc[mi][li][rr] * alpha[cobase + rr] + beta[cobase + rr];
        if (relu) v = fmaxf(v, 0.f);
        yp[rr] = v;   // 8 consecutive channels -> contiguous store
      }
    }
  }
}

// ---------------------------------------------------------------------------
// Direct register-blocked WMMA GEMM (small M, e.g. the wm conv with M=32)
// ---------------------------------------------------------------------------
template <int MT, int LT>
__global__ void __launch_bounds__(256)
gemm_conv_wmma(const __bf16* __restrict__ Xc, const __bf16* __restrict__ Wp,
               const float* __restrict__ alpha, const float* __restrict__ beta,
               float* __restrict__ Y, int L, int M, int K, int relu) {
  const int wave = (blockIdx.x << 3) + (threadIdx.x >> 5);
  const int lane = threadIdx.x & 31;
  const int half = lane >> 4;
  const int lm   = lane & 15;
  const int mtB = M / (16 * MT);
  const int ltB = L / (16 * LT);
  const int tpn = mtB * ltB;
  const int n   = wave / tpn;
  const int r   = wave - n * tpn;
  const int mblk = (r % mtB) * (16 * MT);
  const int lblk = (r / mtB) * (16 * LT);

  const __bf16* wrow[MT];
#pragma unroll
  for (int mi = 0; mi < MT; ++mi)
    wrow[mi] = Wp + (size_t)(mblk + mi * 16 + lm) * K + (half << 3);
  const __bf16* xrow[LT];
#pragma unroll
  for (int li = 0; li < LT; ++li)
    xrow[li] = Xc + ((size_t)n * L + lblk + li * 16 + lm) * K + (half << 3);

  v8f zero = {};
  v8f acc[MT][LT];
#pragma unroll
  for (int mi = 0; mi < MT; ++mi)
#pragma unroll
    for (int li = 0; li < LT; ++li) acc[mi][li] = zero;

  for (int kb = 0; kb < K; kb += 32) {
    v16bf b[LT];
#pragma unroll
    for (int li = 0; li < LT; ++li) {
      v8bf b0 = *(const v8bf*)(xrow[li] + kb);
      v8bf b1 = *(const v8bf*)(xrow[li] + kb + 16);
      if (kb + 32 < K) __builtin_prefetch(xrow[li] + kb + 32, 0, 1); // global_prefetch_b8
#pragma unroll
      for (int j = 0; j < 8; ++j) { b[li][j] = b0[j]; b[li][j + 8] = b1[j]; }
    }
#pragma unroll
    for (int mi = 0; mi < MT; ++mi) {
      v8bf a0 = *(const v8bf*)(wrow[mi] + kb);
      v8bf a1 = *(const v8bf*)(wrow[mi] + kb + 16);
      v16bf a;
#pragma unroll
      for (int j = 0; j < 8; ++j) { a[j] = a0[j]; a[j + 8] = a1[j]; }
#pragma unroll
      for (int li = 0; li < LT; ++li)
        acc[mi][li] = __builtin_amdgcn_wmma_f32_16x16x32_bf16(
            false, a, false, b[li], (short)0, acc[mi][li], false, false);
    }
  }

#pragma unroll
  for (int mi = 0; mi < MT; ++mi) {
    const int cobase = mblk + mi * 16 + (half << 3);
#pragma unroll
    for (int li = 0; li < LT; ++li) {
      const int l = lblk + li * 16 + lm;
      float* yp = Y + ((size_t)n * L + l) * M + cobase;
#pragma unroll
      for (int rr = 0; rr < 8; ++rr) {
        float v = acc[mi][li][rr] * alpha[cobase + rr] + beta[cobase + rr];
        if (relu) v = fmaxf(v, 0.f);
        yp[rr] = v;
      }
    }
  }
}

// ---------------------------------------------------------------------------
// LSH hash: codes -> composite stable-sort keys (code*hL + pos)
// one block per (n,l); rot layout [Cr][nh][nb2]
// ---------------------------------------------------------------------------
__global__ void __launch_bounds__(256)
lsh_hash(const float* __restrict__ xe, const float* __restrict__ rot,
         int* __restrict__ keys, int L, int Cr, int nh, int nb2) {
  __shared__ float sx[128];
  __shared__ float sr[256];
  const int bl = blockIdx.x;
  const int l = bl % L, n = bl / L;
  const int tid = threadIdx.x;
  for (int f = tid; f < Cr; f += 256) sx[f] = xe[((size_t)n * L + l) * Cr + f];
  __syncthreads();
  const int tot = nh * nb2;
  for (int p = tid; p < tot; p += 256) {
    int h = p / nb2, i = p - h * nb2;
    float d = 0.f;
    for (int f = 0; f < Cr; ++f) d += sx[f] * rot[((size_t)f * nh + h) * nb2 + i];
    sr[p] = d;
  }
  __syncthreads();
  if (tid < nh) {
    float mx = -1e30f, mn = 1e30f; int ax = 0, an = 0;
    for (int i = 0; i < nb2; ++i) {
      float v = sr[tid * nb2 + i];
      if (v > mx) { mx = v; ax = i; }
      if (v < mn) { mn = v; an = i; }
    }
    int code = (mx >= -mn) ? ax : (nb2 + an);   // argmax over [rot,-rot]
    code += tid * (2 * nb2);
    int hL = nh * L;
    int pos = tid * L + l;
    keys[(size_t)n * hL + pos] = code * hL + pos;
  }
}

// In-LDS bitonic argsort per batch (hL is a power of two)
__global__ void __launch_bounds__(1024)
bitonic_argsort(const int* __restrict__ keys, int* __restrict__ idxo, int hL) {
  extern __shared__ int sk[];
  const int n = blockIdx.x, tid = threadIdx.x, nt = blockDim.x;
  for (int i = tid; i < hL; i += nt) sk[i] = keys[(size_t)n * hL + i];
  __syncthreads();
  for (int k = 2; k <= hL; k <<= 1)
    for (int j = k >> 1; j > 0; j >>= 1) {
      for (int i = tid; i < hL; i += nt) {
        int ixj = i ^ j;
        if (ixj > i) {
          int a = sk[i], b = sk[ixj];
          bool up = ((i & k) == 0);
          if ((a > b) == up) { sk[i] = b; sk[ixj] = a; }
        }
      }
      __syncthreads();
    }
  for (int i = tid; i < hL; i += nt) idxo[(size_t)n * hL + i] = sk[i] & (hL - 1);
}

// ---------------------------------------------------------------------------
// Chunked attention over sorted buckets (one block per (bucket, h, n)).
// Scatters ret/bs directly by sorted index (no undo array needed).
// ---------------------------------------------------------------------------
__global__ void __launch_bounds__(256)
nlsa_attention(const float* __restrict__ xe, const float* __restrict__ ye,
               const int* __restrict__ idx, float* __restrict__ retb,
               float* __restrict__ bsb, int L, int Cr, int C, int nh, int chunk) {
  __shared__ float xm[96 * 128];
  __shared__ float sc[32 * 96];
  __shared__ float nrm[96];
  __shared__ int   stok[96];
  __shared__ int   sltok[96];
  const int nck = L / chunk;
  const int tc = 3 * chunk;
  const int hL = nh * L;
  const int kblk = blockIdx.x, h = blockIdx.y, n = blockIdx.z;
  const int tid = threadIdx.x;

  if (tid < tc) {  // tokens: [cur, prev, next] buckets
    int d = tid / chunk;
    int r = tid - d * chunk;
    int kb = kblk + ((d == 1) ? -1 : (d == 2) ? 1 : 0);
    kb = (kb + nck) % nck;
    int q = idx[(size_t)n * hL + h * L + kb * chunk + r];
    stok[tid] = q;
    sltok[tid] = q & (L - 1);
  }
  __syncthreads();
  for (int p = tid; p < tc * Cr; p += 256) {
    int j = p / Cr, f = p - j * Cr;
    xm[p] = xe[((size_t)n * L + sltok[j]) * Cr + f];
  }
  __syncthreads();
  if (tid < tc) {
    float s = 0.f;
    for (int f = 0; f < Cr; ++f) { float v = xm[tid * Cr + f]; s += v * v; }
    nrm[tid] = fmaxf(sqrtf(s), 5e-5f);
  }
  __syncthreads();
  for (int p = tid; p < tc * Cr; p += 256) xm[p] /= nrm[p / Cr];
  __syncthreads();
  // raw[i][jj] = (xs_i . xm_jj) = nrm_i * (xm_i . xm_jj)
  for (int p = tid; p < chunk * tc; p += 256) {
    int i = p / tc, jj = p - i * tc;
    float d = 0.f;
    for (int f = 0; f < Cr; ++f) d += xm[i * Cr + f] * xm[jj * Cr + f];
    sc[p] = d * nrm[i];
  }
  __syncthreads();
  if (tid < chunk) {
    float m = -1e30f;
    for (int jj = 0; jj < tc; ++jj) m = fmaxf(m, sc[tid * tc + jj]);
    float s = 0.f;
    for (int jj = 0; jj < tc; ++jj) s += __expf(sc[tid * tc + jj] - m);
    float lse = m + __logf(s);
    for (int jj = 0; jj < tc; ++jj) sc[tid * tc + jj] = __expf(sc[tid * tc + jj] - lse);
    bsb[(size_t)n * hL + stok[tid]] = lse;
  }
  __syncthreads();
  for (int p = tid; p < chunk * C; p += 256) {
    int i = p / C, c = p - i * C;
    float a = 0.f;
    for (int jj = 0; jj < tc; ++jj)
      a += sc[i * tc + jj] * ye[((size_t)n * L + sltok[jj]) * C + c];
    retb[((size_t)n * hL + stok[i]) * C + c] = a;
  }
}

// softmax over hash rounds + residual ; out NLC
__global__ void __launch_bounds__(256)
combine_residual(const float* __restrict__ retb, const float* __restrict__ bsb,
                 const float* __restrict__ xin, float* __restrict__ out,
                 int L, int C, int nh, long total) {
  long t = (long)blockIdx.x * blockDim.x + threadIdx.x;
  if (t >= total) return;
  int c = (int)(t % C);
  long nl = t / C;
  int l = (int)(nl % L);
  int n = (int)(nl / L);
  const int hL = nh * L;
  const float* bp = bsb + (size_t)n * hL + l;
  float m = -1e30f;
  for (int h = 0; h < nh; ++h) m = fmaxf(m, bp[(size_t)h * L]);
  float Z = 0.f, acc = 0.f;
  for (int h = 0; h < nh; ++h) {
    float e = __expf(bp[(size_t)h * L] - m);
    Z += e;
    acc += e * retb[((size_t)n * hL + (size_t)h * L + l) * C + c];
  }
  out[((size_t)n * L + l) * C + c] = acc / Z + xin[((size_t)n * L + l) * C + c];
}

__global__ void __launch_bounds__(256)
maxpool2_nlc(const float* __restrict__ X, float* __restrict__ Y,
             int L, int C, long total) {
  long t = (long)blockIdx.x * blockDim.x + threadIdx.x;
  if (t >= total) return;
  int c = (int)(t % C);
  long nl = t / C;
  int lp = (int)(nl % (L / 2));
  int n  = (int)(nl / (L / 2));
  float a = X[((size_t)n * L + 2 * lp) * C + c];
  float b = X[((size_t)n * L + 2 * lp + 1) * C + c];
  Y[((size_t)n * (L / 2) + lp) * C + c] = fmaxf(a, b);
}

// mean over L then linear [256,512]
__global__ void __launch_bounds__(256)
head_kernel(const float* __restrict__ X, const float* __restrict__ wf,
            const float* __restrict__ bfv, float* __restrict__ out) {
  __shared__ float mean[512];
  const int n = blockIdx.x, tid = threadIdx.x;
  for (int c = tid; c < 512; c += 256) {
    float s = 0.f;
    for (int l = 0; l < 512; ++l) s += X[((size_t)n * 512 + l) * 512 + c];
    mean[c] = s * (1.f / 512.f);
  }
  __syncthreads();
  float s = bfv[tid];
  for (int c = 0; c < 512; ++c) s += mean[c] * wf[(size_t)tid * 512 + c];
  out[n * 256 + tid] = s;
}

// ===========================================================================
extern "C" void kernel_launch(void* const* d_in, const int* in_sizes, int n_in,
                              void* d_out, int out_size, void* d_ws, size_t ws_size,
                              hipStream_t stream) {
  (void)in_sizes; (void)n_in; (void)out_size; (void)ws_size;
  int ii = 0;
  const float* x   = (const float*)d_in[ii++];
  const float* w0  = (const float*)d_in[ii++];
  const float* b0  = (const float*)d_in[ii++];
  const float* g0  = (const float*)d_in[ii++];
  const float* be0 = (const float*)d_in[ii++];
  const float* mu0 = (const float*)d_in[ii++];
  const float* vr0 = (const float*)d_in[ii++];
  const float* P[3][17];
  for (int i = 0; i < 3; ++i)
    for (int j = 0; j < 17; ++j) P[i][j] = (const float*)d_in[ii++];
  // per block order: w1,b1,g1,be1,mu1,var1, wm,bm, wa,ba, rot, w2,b2,g2,be2,mu2,var2
  const float* wf  = (const float*)d_in[ii++];
  const float* bfv = (const float*)d_in[ii++];

  char* wsb = (char*)d_ws;
  size_t off = 0;
  auto carve = [&](size_t bytes) -> void* {
    off = (off + 255) & ~(size_t)255;
    void* p = wsb + off;
    off += bytes;
    return p;
  };
  float*          actA  = (float*)carve(33554432);            // [32][2048][128] max
  float*          actB  = (float*)carve(33554432);
  unsigned short* XCOL  = (unsigned short*)carve(50331648);   // [32][L][3*cout] bf16
  unsigned short* WP    = (unsigned short*)carve(1572864);    // [512][1536] bf16
  float*          ALPHA = (float*)carve(512 * 4);
  float*          BETA  = (float*)carve(512 * 4);
  float*          XE    = (float*)carve(8388608);             // [32][L][Cr]
  float*          YE    = (float*)carve(33554432);            // [32][L][cout]
  int*            KEYS  = (int*)carve(1048576);
  int*            IDX   = (int*)carve(1048576);
  float*          RETB  = (float*)carve(134217728);           // [32][hL][cout]
  float*          BSB   = (float*)carve(1048576);

  auto gr = [](long t) { return (unsigned)((t + 255) / 256); };
  auto gemm = [&](const unsigned short* xc, const unsigned short* wp,
                  float* y, int L, int M, int K, int relu) {
    if (M % 128 == 0) {
      int blocks = 32 * (L / 32);   // one workgroup per (n, 32-wide l panel)
      switch (M / 128) {
        case 1: gemm_conv_wmma_tdm<1><<<blocks, 256, 0, stream>>>(
                    (const __bf16*)xc, (const __bf16*)wp, ALPHA, BETA, y, L, M, K, relu);
                break;
        case 2: gemm_conv_wmma_tdm<2><<<blocks, 256, 0, stream>>>(
                    (const __bf16*)xc, (const __bf16*)wp, ALPHA, BETA, y, L, M, K, relu);
                break;
        default: gemm_conv_wmma_tdm<4><<<blocks, 256, 0, stream>>>(
                    (const __bf16*)xc, (const __bf16*)wp, ALPHA, BETA, y, L, M, K, relu);
                break;
      }
    } else {
      int waves = 32 * (M / 32) * (L / 32);
      gemm_conv_wmma<2, 2><<<waves / 8, 256, 0, stream>>>(
          (const __bf16*)xc, (const __bf16*)wp, ALPHA, BETA, y, L, M, K, relu);
    }
  };

  // stem
  long t0 = 32L * 2048 * 64;
  conv0_bn_relu_pool<<<gr(t0), 256, 0, stream>>>(x, w0, b0, g0, be0, mu0, vr0, actA, t0);

  const int NH[3] = {4, 4, 2}, CHK[3] = {8, 16, 32}, NB2[3] = {64, 32, 8};
  float* act = actA;
  for (int i = 0; i < 3; ++i) {
    const int cin = 64 << i, cout = 128 << i, L = 2048 >> i;
    const int Cr = cout / 4, nh = NH[i], chunk = CHK[i], nb2 = NB2[i];
    const int hL = nh * L;

    // conv w1 + bn1 + relu -> x1 (actB)
    long ti1 = 32L * L * cin;
    im2col3<<<gr(ti1), 256, 0, stream>>>(act, XCOL, L, cin, ti1);
    long tw1 = (long)cout * cin * 3;
    pack_weights<<<gr(tw1), 256, 0, stream>>>(P[i][0], WP, cout, cin, 3, tw1);
    make_affine<<<gr(cout), 256, 0, stream>>>(P[i][1], P[i][2], P[i][3], P[i][4], P[i][5],
                                              ALPHA, BETA, cout, 1);
    gemm(XCOL, WP, actB, L, cout, 3 * cin, 1);
    float* x1 = actB;

    // shared im2col of x1; wm (K=3) and wa (K=1 zero-padded into K=3 slots)
    long ti2 = 32L * L * cout;
    im2col3<<<gr(ti2), 256, 0, stream>>>(x1, XCOL, L, cout, ti2);
    long twm = (long)Cr * cout * 3;
    pack_weights<<<gr(twm), 256, 0, stream>>>(P[i][6], WP, Cr, cout, 3, twm);
    make_affine<<<gr(Cr), 256, 0, stream>>>(P[i][7], nullptr, nullptr, nullptr, nullptr,
                                            ALPHA, BETA, Cr, 0);
    gemm(XCOL, WP, XE, L, Cr, 3 * cout, 0);
    long twa = (long)cout * cout * 3;
    pack_weights<<<gr(twa), 256, 0, stream>>>(P[i][8], WP, cout, cout, 1, twa);
    make_affine<<<gr(cout), 256, 0, stream>>>(P[i][9], nullptr, nullptr, nullptr, nullptr,
                                              ALPHA, BETA, cout, 0);
    gemm(XCOL, WP, YE, L, cout, 3 * cout, 0);

    // LSH hash + stable bitonic argsort
    lsh_hash<<<32 * L, 256, 0, stream>>>(XE, P[i][10], KEYS, L, Cr, nh, nb2);
    bitonic_argsort<<<32, 1024, hL * 4, stream>>>(KEYS, IDX, hL);

    // bucketed attention, combine across hash rounds, residual -> actA
    dim3 ag(L / chunk, nh, 32);
    nlsa_attention<<<ag, 256, 0, stream>>>(XE, YE, IDX, RETB, BSB, L, Cr, cout, nh, chunk);
    combine_residual<<<gr(ti2), 256, 0, stream>>>(RETB, BSB, x1, actA, L, cout, nh, ti2);

    // conv w2 + bn2 + relu -> actB
    im2col3<<<gr(ti2), 256, 0, stream>>>(actA, XCOL, L, cout, ti2);
    long tw2 = (long)cout * cout * 3;
    pack_weights<<<gr(tw2), 256, 0, stream>>>(P[i][11], WP, cout, cout, 3, tw2);
    make_affine<<<gr(cout), 256, 0, stream>>>(P[i][12], P[i][13], P[i][14], P[i][15], P[i][16],
                                              ALPHA, BETA, cout, 1);
    gemm(XCOL, WP, actB, L, cout, 3 * cout, 1);

    if (i < 2) {
      long tp = 32L * (L / 2) * cout;
      maxpool2_nlc<<<gr(tp), 256, 0, stream>>>(actB, actA, L, cout, tp);
      act = actA;
    } else {
      act = actB;
    }
  }

  head_kernel<<<32, 256, 0, stream>>>(act, wf, bfv, (float*)d_out);
}
